// ARMAGNNConv_33895881900097
// MI455X (gfx1250) — matrix operand order repaired and checked by reading, not compiled, not run on hardware
//
#include <hip/hip_runtime.h>
#include <hip/hip_bf16.h>
#include <stdint.h>

#define IN_CH  512
#define OUT_CH 256

typedef __attribute__((ext_vector_type(16))) __bf16 v16bf;
typedef __attribute__((ext_vector_type(8)))  float  v8f;

__device__ __forceinline__ __bf16 f2bf(float f) { return (__bf16)f; }

// ---------------- utility kernels ----------------

__global__ void k_zero(float* __restrict__ p, int n) {
    int i = blockIdx.x * blockDim.x + threadIdx.x;
    if (i < n) p[i] = 0.0f;
}

__global__ void k_deg(const long long* __restrict__ ei, float* __restrict__ deg, int E) {
    int e = blockIdx.x * blockDim.x + threadIdx.x;
    if (e < E) atomicAdd(&deg[(int)ei[e]], 1.0f);
}

__global__ void k_dis(float* __restrict__ deg, int n) {
    int i = blockIdx.x * blockDim.x + threadIdx.x;
    if (i < n) {
        float d = deg[i];
        deg[i] = (d > 0.0f) ? rsqrtf(d) : 0.0f;   // in-place: deg -> deg^{-1/2}
    }
}

// out[i] += weights[k] * t[i]
__global__ void k_axpy(float* __restrict__ out, const float* __restrict__ t,
                       const float* __restrict__ wts, int k, int n) {
    int i = blockIdx.x * blockDim.x + threadIdx.x;
    if (i < n) out[i] += wts[k] * t[i];
}

// ---------------- WMMA GEMM: out = x @ W^T + b ----------------
// x: [N, 512] f32 row-major,  W: [256, 512] f32 row-major (so B[k][n] = W[n][k],
// i.e. each lane reads K-contiguous floats of one W row -> fully coalesced-ish).
// One wave: 16 rows x 64 cols (4 accumulators). Block: 8 waves -> 128 rows.
__global__ __launch_bounds__(256) void k_gemm(const float* __restrict__ x,
                                              const float* __restrict__ w,
                                              const float* __restrict__ bias,
                                              float* __restrict__ out, int nrows) {
    const int lane = threadIdx.x & 31;
    const int wave = threadIdx.x >> 5;
    const int half = lane >> 4;       // 0 or 1
    const int l16  = lane & 15;
    const int m0 = blockIdx.x * 128 + wave * 16;
    const int n0 = blockIdx.y * 64;

    v8f acc0 = {}; v8f acc1 = {}; v8f acc2 = {}; v8f acc3 = {};

    int mrow = m0 + l16;
    if (mrow >= nrows) mrow = nrows - 1;          // clamp loads; store is masked
    const float* xrow = x + (size_t)mrow * IN_CH;
    const float* wr0 = w + (size_t)(n0 +  0 + l16) * IN_CH;
    const float* wr1 = w + (size_t)(n0 + 16 + l16) * IN_CH;
    const float* wr2 = w + (size_t)(n0 + 32 + l16) * IN_CH;
    const float* wr3 = w + (size_t)(n0 + 48 + l16) * IN_CH;

    for (int kb = 0; kb < IN_CH; kb += 32) {
        // A fragment: 16-bit A 16x32 layout (ISA 7.12.2):
        // VGPR v holds K = (v>>2)*16 + (v&3)*2 + 8*half  (+0,+1)
        v16bf a;
#pragma unroll
        for (int v = 0; v < 8; ++v) {
            int k = kb + ((v >> 2) << 4) + ((v & 3) << 1) + (half << 3);
            a[2 * v]     = f2bf(xrow[k]);
            a[2 * v + 1] = f2bf(xrow[k + 1]);
        }
        // B fragments: same K order as A -> VGPR v holds
        // K = (v>>2)*16 + (v&3)*2 + 8*half (+0,+1), N = l16.
        v16bf b0, b1, b2, b3;
#pragma unroll
        for (int v = 0; v < 8; ++v) {
            int k = kb + ((v >> 2) << 4) + ((v & 3) << 1) + (half << 3);
            b0[2 * v] = f2bf(wr0[k]); b0[2 * v + 1] = f2bf(wr0[k + 1]);
            b1[2 * v] = f2bf(wr1[k]); b1[2 * v + 1] = f2bf(wr1[k + 1]);
            b2[2 * v] = f2bf(wr2[k]); b2[2 * v + 1] = f2bf(wr2[k + 1]);
            b3[2 * v] = f2bf(wr3[k]); b3[2 * v + 1] = f2bf(wr3[k + 1]);
        }
        acc0 = __builtin_amdgcn_wmma_f32_16x16x32_bf16(false, a, false, b0, (short)0, acc0, false, false);
        acc1 = __builtin_amdgcn_wmma_f32_16x16x32_bf16(false, a, false, b1, (short)0, acc1, false, false);
        acc2 = __builtin_amdgcn_wmma_f32_16x16x32_bf16(false, a, false, b2, (short)0, acc2, false, false);
        acc3 = __builtin_amdgcn_wmma_f32_16x16x32_bf16(false, a, false, b3, (short)0, acc3, false, false);
    }

    // C/D layout: VGPR r, lane l -> M = r + 8*half, N = l16. Add bias, masked store.
    v8f accs[4] = {acc0, acc1, acc2, acc3};
#pragma unroll
    for (int t = 0; t < 4; ++t) {
        int n = n0 + t * 16 + l16;
        float bv = bias[n];
#pragma unroll
        for (int r = 0; r < 8; ++r) {
            int m = m0 + r + 8 * half;
            if (m < nrows) out[(size_t)m * OUT_CH + n] = accs[t][r] + bv;
        }
    }
}

// ---------------- SpMM: y[row[e]] += dis[row]*dis[col] * h[col[e]] ----------------
// Edge-parallel; one thread = one edge x 4 channels (float4 gather, 4 f32 atomics).
// h (10 MB) and y (10 MB) are L2-resident on MI455X (192 MB L2).
__global__ __launch_bounds__(256) void k_spmm(const float* __restrict__ h,
                                              float* __restrict__ y,
                                              const long long* __restrict__ ei,
                                              const float* __restrict__ dis, int E) {
    long long tid = (long long)blockIdx.x * blockDim.x + threadIdx.x;
    int e = (int)(tid >> 6);                    // 64 threads per edge (256 ch / 4)
    if (e >= E) return;
    int cq = ((int)tid & 63) << 2;
    int r = (int)ei[e];
    int c = (int)ei[(size_t)E + e];
    float nrm = dis[r] * dis[c];
    const float4 hv = *(const float4*)(h + (size_t)c * OUT_CH + cq);
    float* yp = y + (size_t)r * OUT_CH + cq;
    atomicAdd(yp + 0, nrm * hv.x);
    atomicAdd(yp + 1, nrm * hv.y);
    atomicAdd(yp + 2, nrm * hv.z);
    atomicAdd(yp + 3, nrm * hv.w);
}

// ---------------- driver ----------------

extern "C" void kernel_launch(void* const* d_in, const int* in_sizes, int n_in,
                              void* d_out, int out_size, void* d_ws, size_t ws_size,
                              hipStream_t stream) {
    const float*     x   = (const float*)d_in[0];
    const long long* ei  = (const long long*)d_in[1];   // int64 [2, E]
    const float*     lw  = (const float*)d_in[2];
    const float*     lb  = (const float*)d_in[3];
    const float*     wts = (const float*)d_in[4];
    float* out = (float*)d_out;

    const int N = in_sizes[0] / IN_CH;
    const int E = in_sizes[1] / 2;
    const int NC = N * OUT_CH;

    float* ws   = (float*)d_ws;
    int Npad    = ((N + 255) / 256) * 256;
    float* dis  = ws;                        // N floats (deg -> deg^{-1/2})
    float* tmp1 = ws + Npad;                 // N*256 floats
    float* tmp2 = tmp1 + (size_t)NC;         // N*256 floats

    const int TB = 256;
    int nb_N  = (N  + TB - 1) / TB;
    int nb_E  = (E  + TB - 1) / TB;
    int nb_NC = (NC + TB - 1) / TB;
    long long spmm_threads = (long long)E * 64;
    int nb_spmm = (int)((spmm_threads + TB - 1) / TB);

    // degree -> deg^{-1/2}
    k_zero<<<nb_N, TB, 0, stream>>>(dis, N);
    k_deg<<<nb_E, TB, 0, stream>>>(ei, dis, E);
    k_dis<<<nb_N, TB, 0, stream>>>(dis, N);

    // out = x @ W^T + b  (bf16 WMMA, f32 accumulate)
    dim3 gg((N + 127) / 128, OUT_CH / 64);
    k_gemm<<<gg, TB, 0, stream>>>(x, lw, lb, out, N);

    // k = 0: out += w0 * spmm(out)
    k_zero<<<nb_NC, TB, 0, stream>>>(tmp1, NC);
    k_spmm<<<nb_spmm, TB, 0, stream>>>(out, tmp1, ei, dis, E);
    k_axpy<<<nb_NC, TB, 0, stream>>>(out, tmp1, wts, 0, NC);

    // k = 1: out += w1 * spmm(spmm(out))
    k_zero<<<nb_NC, TB, 0, stream>>>(tmp1, NC);
    k_spmm<<<nb_spmm, TB, 0, stream>>>(out, tmp1, ei, dis, E);
    k_zero<<<nb_NC, TB, 0, stream>>>(tmp2, NC);
    k_spmm<<<nb_spmm, TB, 0, stream>>>(tmp1, tmp2, ei, dis, E);
    k_axpy<<<nb_NC, TB, 0, stream>>>(out, tmp2, wts, 1, NC);
}
